// ScaledDotProductAttention_80204219286039
// MI455X (gfx1250) — compile-verified, matrix operand order checked by reading, not run on hardware
//
#include <hip/hip_runtime.h>

typedef __attribute__((ext_vector_type(16))) _Float16 v16h;
typedef __attribute__((ext_vector_type(8)))  float    v8f;

constexpr int Bn = 8, Ln = 2048, Dn = 512;
constexpr int QROW = 520;   // padded f16 elements per Q row (bank-conflict-free)
constexpr int SROW = 2052;  // padded f32 elements per score row (stride%64 dwords = 4)
constexpr float SCALE = 0.04419417382415922f;  // 1/sqrt(512)

// Pack two f32 into two f16 (round-toward-zero) as one dword.
static __device__ __forceinline__ uint32_t pkf16(float a, float b) {
  auto t = __builtin_amdgcn_cvt_pkrtz(a, b);   // __fp16 ext_vector(2)
  return __builtin_bit_cast(uint32_t, t);
}

// K-offset inside a 16-bit WMMA A/B fragment for VGPR r, lane-half h.
// Layout per CDNA5 ISA 7.12.2: VGPR0..3 hold K = h*8 + {0,1 / 2,3 / 4,5 / 6,7},
// VGPR4..7 hold K = 16 + h*8 + {0,1 / 2,3 / 4,5 / 6,7}.
static __device__ __forceinline__ int frag_kbase(int r, int h) {
  return ((r & 4) << 2) + (h << 3) + ((r & 3) << 1);
}

extern "C" __global__ __launch_bounds__(256)
void attn_fused(const float* __restrict__ q, const float* __restrict__ kmat,
                const float* __restrict__ vmat, float* __restrict__ out,
                float* __restrict__ attn)
{
  extern __shared__ char smem[];
  _Float16* Qh  = (_Float16*)smem;                                    // 16*QROW f16
  float*    S   = (float*)(smem + 16 * QROW * sizeof(_Float16));      // 16*SROW f32
  float*    red = S + 16 * SROW;                                      // 256 f32
  float*    red2 = red + 256;                                         // 16 f32

  const int tid  = threadIdx.x;
  const int wave = tid >> 5;
  const int lane = tid & 31;
  const int m16  = lane & 15;      // row (A) or column (B/C) index inside tile
  const int half = lane >> 4;      // lane half selects K/M sub-block
  const int b    = blockIdx.y;
  const int i0   = blockIdx.x * 16;

  const float* qb = q    + ((size_t)b * Ln + i0) * Dn;
  const float* kb = kmat + (size_t)b * Ln * Dn;
  const float* vb = vmat + (size_t)b * Ln * Dn;

  // ---- Stage Q tile (16 x 512) into LDS as f16 ----
  for (int idx = tid; idx < 16 * (Dn / 2); idx += 256) {
    int row = idx / (Dn / 2);
    int dp  = (idx % (Dn / 2)) * 2;
    *(uint32_t*)(&Qh[row * QROW + dp]) = pkf16(qb[row * Dn + dp], qb[row * Dn + dp + 1]);
  }
  __syncthreads();

  // ---- Pass 1: S = (Q K^T) * scale, tiles of 16x16, K-dim = D ----
  // Q A-fragments are jt-invariant: hoist all 16 (kc chunks of 32) into registers.
  v16h afrag[16];
#pragma unroll
  for (int kt = 0; kt < 16; ++kt) {
    uint32_t* ap = (uint32_t*)&afrag[kt];
#pragma unroll
    for (int r = 0; r < 8; ++r)
      ap[r] = *(const uint32_t*)(&Qh[m16 * QROW + kt * 32 + frag_kbase(r, half)]);
  }

  for (int jt = wave; jt < Ln / 16; jt += 8) {
    v8f acc = {};
    const float* kcol = kb + (size_t)(jt * 16 + m16) * Dn;  // B column n = m16
#pragma unroll
    for (int kt = 0; kt < 16; ++kt) {
      v16h bf;
      uint32_t* bp = (uint32_t*)&bf;
#pragma unroll
      for (int r = 0; r < 8; ++r) {
        const int kk = kt * 32 + frag_kbase(r, half);
        bp[r] = pkf16(kcol[kk], kcol[kk + 1]);
      }
      acc = __builtin_amdgcn_wmma_f32_16x16x32_f16(
          false, afrag[kt], false, bf, (short)0, acc, false, false);
    }
#pragma unroll
    for (int r = 0; r < 8; ++r)
      S[(half * 8 + r) * SROW + jt * 16 + m16] = acc[r] * SCALE;  // C: M=half*8+r, N=m16
  }
  __syncthreads();

  // ---- Pass 2: exact row softmax over 2048 columns, 16 threads per row ----
  const int row = tid >> 4;
  const int sub = tid & 15;
  float mx = -3.402823466e+38f;
  for (int j = sub; j < Ln; j += 16) mx = fmaxf(mx, S[row * SROW + j]);
  red[row * 16 + sub] = mx;
  __syncthreads();
  if (sub == 0) {
    float m2 = red[row * 16];
    for (int t = 1; t < 16; ++t) m2 = fmaxf(m2, red[row * 16 + t]);
    red2[row] = m2;
  }
  __syncthreads();
  const float rmax = red2[row];
  float sum = 0.0f;
  for (int j = sub; j < Ln; j += 16) {
    float e = __expf(S[row * SROW + j] - rmax);
    S[row * SROW + j] = e;
    sum += e;
  }
  red[row * 16 + sub] = sum;
  __syncthreads();
  if (sub == 0) {
    float s2 = 0.0f;
    for (int t = 0; t < 16; ++t) s2 += red[row * 16 + t];
    red2[row] = s2;
  }
  __syncthreads();
  const float inv = 1.0f / red2[row];
  float* arow = attn + ((size_t)b * Ln + i0 + row) * Ln;
  for (int j = sub; j < Ln; j += 16) {   // adjacent lanes hit adjacent j -> coalesced
    float p = S[row * SROW + j] * inv;
    S[row * SROW + j] = p;
    arow[j] = p;
  }
  __syncthreads();

  // ---- Pass 3: out = P x V, tiles of 16x16 over D, K-dim = L ----
  // Outer loop over K-chunks so the P A-fragment is built once and reused by
  // all 4 d-tiles this wave owns (nt = wave + 8*t).
  {
    v8f acc[4] = {{}, {}, {}, {}};
    for (int jc = 0; jc < Ln; jc += 32) {
      v16h a;
      uint32_t* ap = (uint32_t*)&a;
#pragma unroll
      for (int r = 0; r < 8; ++r) {
        const int kk = jc + frag_kbase(r, half);
        ap[r] = pkf16(S[m16 * SROW + kk], S[m16 * SROW + kk + 1]);
      }
#pragma unroll
      for (int t = 0; t < 4; ++t) {
        const float* vcol = vb + (wave + 8 * t) * 16 + m16;  // B column n = m16
        v16h bf;
        uint32_t* bp = (uint32_t*)&bf;
#pragma unroll
        for (int r = 0; r < 8; ++r) {
          const int kk = jc + frag_kbase(r, half);
          bp[r] = pkf16(vcol[(size_t)kk * Dn], vcol[(size_t)(kk + 1) * Dn]);
        }
        acc[t] = __builtin_amdgcn_wmma_f32_16x16x32_f16(
            false, a, false, bf, (short)0, acc[t], false, false);
      }
    }
#pragma unroll
    for (int t = 0; t < 4; ++t) {
      float* orow = out + ((size_t)b * Ln + i0) * Dn + (wave + 8 * t) * 16 + m16;
#pragma unroll
      for (int r = 0; r < 8; ++r)
        orow[(size_t)(half * 8 + r) * Dn] = acc[t][r];
    }
  }
}

extern "C" void kernel_launch(void* const* d_in, const int* in_sizes, int n_in,
                              void* d_out, int out_size, void* d_ws, size_t ws_size,
                              hipStream_t stream) {
  const float* q = (const float*)d_in[0];
  const float* k = (const float*)d_in[1];
  const float* v = (const float*)d_in[2];
  // d_in[3] = mask, ignored by the reference computation.
  float* out  = (float*)d_out;
  float* attn = out + (size_t)Bn * Ln * Dn;

  const size_t shmem = (size_t)16 * QROW * sizeof(_Float16)
                     + (size_t)(16 * SROW + 256 + 16) * sizeof(float);
  dim3 grid(Ln / 16, Bn);
  dim3 block(256);
  hipLaunchKernelGGL(attn_fused, grid, block, shmem, stream, q, k, v, out, attn);
}
